// MEI_60507499266614
// MI455X (gfx1250) — compile-verified, matrix operand order checked by reading, not run on hardware
//
#include <hip/hip_runtime.h>
#include <math.h>

// ---------------------------------------------------------------------------
// Types for WMMA
// ---------------------------------------------------------------------------
typedef _Float16 h8   __attribute__((ext_vector_type(8)));
typedef _Float16 v16h __attribute__((ext_vector_type(16)));
typedef float    f32x8 __attribute__((ext_vector_type(8)));

// Problem constants
#define Bb   16
#define Nn   4096
#define Dd   512
#define Hh   2048
#define GIN_ 4096
#define PIN_ 1024
#define MROWS (Bb * Nn)          // 65536 path-token rows

// GEMM tiling
#define BM 64
#define BN 128
#define BK 32
#define LDA (BK + 8)             // 40 halfs = 80 bytes (16B multiple)
#define LDB (BK + 8)

// ---------------------------------------------------------------------------
// Row stats: mean / rstd (biased var, eps=1e-5) for rows of length L
// ---------------------------------------------------------------------------
__global__ __launch_bounds__(256) void k_rowstats(const float* __restrict__ x,
                                                  int L, float* __restrict__ mu,
                                                  float* __restrict__ rs) {
  int row = blockIdx.x, tid = threadIdx.x;
  __shared__ float s1[256], s2[256];
  float a = 0.f, b = 0.f;
  const float* xr = x + (size_t)row * L;
  for (int i = tid; i < L; i += 256) { float v = xr[i]; a += v; b += v * v; }
  s1[tid] = a; s2[tid] = b; __syncthreads();
  for (int s = 128; s > 0; s >>= 1) {
    if (tid < s) { s1[tid] += s1[tid + s]; s2[tid] += s2[tid + s]; }
    __syncthreads();
  }
  if (tid == 0) {
    float m = s1[0] / (float)L;
    float v = s2[0] / (float)L - m * m;
    mu[row] = m;
    rs[row] = rsqrtf(fmaxf(v, 0.f) + 1e-5f);
  }
}

// ---------------------------------------------------------------------------
// path_W [K=1024][N=512] fp32 -> WhT [N=512][K=1024] f16
// ---------------------------------------------------------------------------
__global__ __launch_bounds__(256) void k_cvt_w(const float* __restrict__ W,
                                               _Float16* __restrict__ WhT) {
  int id = blockIdx.x * 256 + threadIdx.x;   // 512*1024 total
  int n = id >> 10, k = id & 1023;
  WhT[id] = (_Float16)W[(size_t)k * Dd + n];
}

// ---------------------------------------------------------------------------
// Main GEMM: Xp[65536,512] = (LN(F_path)*g+b) @ path_W + path_b  (f16 WMMA)
// Software-pipelined: global prefetch of step k+1 overlaps LDS loads + WMMA
// of step k; all fragment ds_loads hoisted before the 4-WMMA chain.
// ---------------------------------------------------------------------------
__global__ __launch_bounds__(256) void k_path_gemm(
    const float* __restrict__ Fp,    // [65536,1024]
    const float* __restrict__ mu, const float* __restrict__ rs,
    const float* __restrict__ lng, const float* __restrict__ lnb,   // [1024]
    const _Float16* __restrict__ WhT,  // [512,1024] f16
    const float* __restrict__ bias,    // [512]
    float* __restrict__ Xp)            // [65536,512]
{
  __shared__ __align__(16) _Float16 As[BM * LDA];
  __shared__ __align__(16) _Float16 Bs[BN * LDB];

  const int tid  = threadIdx.x;
  const int m0   = blockIdx.y * BM;
  const int n0   = blockIdx.x * BN;
  const int wave = tid >> 5, lane = tid & 31;
  const int row  = lane & 15, hi = lane >> 4;
  const int msub  = (wave & 3) * 16;      // 4 M-subtiles
  const int nbase = (wave >> 2) * 64;     // 2 wave groups x 64 cols

  f32x8 acc0 = {}, acc1 = {}, acc2 = {}, acc3 = {};

  // A stage mapping: each thread loads 8 floats of one row
  const int ar = tid >> 2, ac = (tid & 3) * 8;
  const int am = m0 + ar;
  const float amu = mu[am], ars = rs[am];
  // B stage mapping: each thread copies 16 halfs (32B)
  const int br = tid >> 1, bc = (tid & 1) * 16;

  const float*    aSrc = Fp + (size_t)am * PIN_ + ac;
  const _Float16* bSrc = WhT + (size_t)(n0 + br) * PIN_ + bc;
  const float*    lgp  = lng + ac;
  const float*    lbp  = lnb + ac;

  union Frag { v16h v; h8 p[2]; };

  // ---- pipeline prologue: prefetch K-step 0 into registers ----
  float4 ar0 = *(const float4*)(aSrc + 0);
  float4 ar1 = *(const float4*)(aSrc + 4);
  float4 br0 = ((const float4*)bSrc)[0];
  float4 br1 = ((const float4*)bSrc)[1];
  float4 gr0 = *(const float4*)(lgp + 0);
  float4 gr1 = *(const float4*)(lgp + 4);
  float4 nr0 = *(const float4*)(lbp + 0);
  float4 nr1 = *(const float4*)(lbp + 4);

  for (int k0 = 0; k0 < PIN_; k0 += BK) {
    // ---- stage prefetched registers into LDS (LN+affine+f16 for A) ----
    {
      float xs[8] = {ar0.x, ar0.y, ar0.z, ar0.w, ar1.x, ar1.y, ar1.z, ar1.w};
      float gs[8] = {gr0.x, gr0.y, gr0.z, gr0.w, gr1.x, gr1.y, gr1.z, gr1.w};
      float bs[8] = {nr0.x, nr0.y, nr0.z, nr0.w, nr1.x, nr1.y, nr1.z, nr1.w};
      _Float16* dst = &As[ar * LDA + ac];
#pragma unroll
      for (int i = 0; i < 8; ++i)
        dst[i] = (_Float16)(((xs[i] - amu) * ars) * gs[i] + bs[i]);
      float4* bd = (float4*)&Bs[br * LDB + bc];
      bd[0] = br0;
      bd[1] = br1;
    }
    __syncthreads();

    // ---- issue global prefetch for step k0+BK (overlaps compute below) ----
    const int kn = k0 + BK;
    if (kn < PIN_) {
      ar0 = *(const float4*)(aSrc + kn);
      ar1 = *(const float4*)(aSrc + kn + 4);
      br0 = ((const float4*)(bSrc + kn))[0];
      br1 = ((const float4*)(bSrc + kn))[1];
      gr0 = *(const float4*)(lgp + kn);
      gr1 = *(const float4*)(lgp + kn + 4);
      nr0 = *(const float4*)(lbp + kn);
      nr1 = *(const float4*)(lbp + kn + 4);
    }

    // ---- load ALL fragments (ISA 16-bit lane layout), then WMMA chain ----
    Frag a, f0, f1, f2, f3;
    const _Float16* ap = &As[(msub + row) * LDA + hi * 8];
    a.p[0] = *(const h8*)ap;           // K = hi*8 .. +7
    a.p[1] = *(const h8*)(ap + 16);    // K = 16+hi*8 .. +7
    const _Float16* bp = &Bs[(nbase + row) * LDB + hi * 8];
    f0.p[0] = *(const h8*)bp;
    f0.p[1] = *(const h8*)(bp + 16);
    f1.p[0] = *(const h8*)(bp + 16 * LDB);
    f1.p[1] = *(const h8*)(bp + 16 * LDB + 16);
    f2.p[0] = *(const h8*)(bp + 32 * LDB);
    f2.p[1] = *(const h8*)(bp + 32 * LDB + 16);
    f3.p[0] = *(const h8*)(bp + 48 * LDB);
    f3.p[1] = *(const h8*)(bp + 48 * LDB + 16);

    acc0 = __builtin_amdgcn_wmma_f32_16x16x32_f16(false, a.v, false, f0.v,
                                                  (short)0, acc0, false, false);
    acc1 = __builtin_amdgcn_wmma_f32_16x16x32_f16(false, a.v, false, f1.v,
                                                  (short)0, acc1, false, false);
    acc2 = __builtin_amdgcn_wmma_f32_16x16x32_f16(false, a.v, false, f2.v,
                                                  (short)0, acc2, false, false);
    acc3 = __builtin_amdgcn_wmma_f32_16x16x32_f16(false, a.v, false, f3.v,
                                                  (short)0, acc3, false, false);
    __syncthreads();
  }

  // ---- epilogue: D layout — lanes 0-15: M=r, lanes 16-31: M=8+r; N=row ----
  const int gm = m0 + msub + hi * 8;
  const int gn = n0 + nbase + row;
#pragma unroll
  for (int r = 0; r < 8; ++r) {
    float* orow = &Xp[(size_t)(gm + r) * Dd + gn];
    orow[0]  = acc0[r] + bias[gn];
    orow[16] = acc1[r] + bias[gn + 16];
    orow[32] = acc2[r] + bias[gn + 32];
    orow[48] = acc3[r] + bias[gn + 48];
  }
}

// ---------------------------------------------------------------------------
// Generic LN projection: out[b,d] = sum_k ((x[b,k]-mu)*rs*g[k]+bb[k])*W[k,d] + bias[d]
// ---------------------------------------------------------------------------
__global__ __launch_bounds__(256) void k_ln_proj(
    const float* __restrict__ x, const float* __restrict__ mu,
    const float* __restrict__ rs, const float* __restrict__ lng,
    const float* __restrict__ lnb, const float* __restrict__ W,
    const float* __restrict__ bias, float* __restrict__ out,
    int L, int Dout)
{
  int id = blockIdx.x * 256 + threadIdx.x;
  int d = id % Dout, b = id / Dout;
  float m = mu[b], r = rs[b], acc = 0.f;
  const float* xr = x + (size_t)b * L;
  for (int k = 0; k < L; ++k) {
    float xn = (xr[k] - m) * r * lng[k] + lnb[k];
    acc += xn * W[(size_t)k * Dout + d];
  }
  out[id] = acc + bias[d];
}

// ---------------------------------------------------------------------------
// Softmax gate: w[b,e] = softmax_e( LN(x[b,:])*g+bb @ W[L,E] + bias )
// ---------------------------------------------------------------------------
__global__ __launch_bounds__(256) void k_gate(
    const float* __restrict__ x, const float* __restrict__ mu,
    const float* __restrict__ rs, const float* __restrict__ lng,
    const float* __restrict__ lnb, const float* __restrict__ W,
    const float* __restrict__ bias, float* __restrict__ out, int L, int E)
{
  int b = blockIdx.x, tid = threadIdx.x;
  __shared__ float red[4 * 256];
  float m = mu[b], r = rs[b];
  float acc[4] = {0.f, 0.f, 0.f, 0.f};
  const float* xr = x + (size_t)b * L;
  for (int d = tid; d < L; d += 256) {
    float xn = (xr[d] - m) * r * lng[d] + lnb[d];
    for (int e = 0; e < E; ++e) acc[e] += xn * W[(size_t)d * E + e];
  }
  for (int e = 0; e < 4; ++e) red[e * 256 + tid] = acc[e];
  __syncthreads();
  for (int s = 128; s > 0; s >>= 1) {
    if (tid < s)
      for (int e = 0; e < 4; ++e) red[e * 256 + tid] += red[e * 256 + tid + s];
    __syncthreads();
  }
  if (tid == 0) {
    float lg[4]; float mx = -1e30f;
    for (int e = 0; e < E; ++e) { lg[e] = red[e * 256] + bias[e]; mx = fmaxf(mx, lg[e]); }
    float s = 0.f;
    for (int e = 0; e < E; ++e) { lg[e] = __expf(lg[e] - mx); s += lg[e]; }
    for (int e = 0; e < E; ++e) out[b * E + e] = lg[e] / s;
  }
}

// ---------------------------------------------------------------------------
// Expert MLP up: hbuf[e,b,j] = gelu( (LN(x)*g_e+b_e) @ W1[e] + b1[e] )
// ---------------------------------------------------------------------------
__global__ __launch_bounds__(256) void k_expert_h(
    const float* __restrict__ x, const float* __restrict__ mu,
    const float* __restrict__ rs, const float* __restrict__ lng,
    const float* __restrict__ lnb, const float* __restrict__ W1,
    const float* __restrict__ b1, float* __restrict__ hbuf,
    int E, int B, int D, int H, int perExpertX)
{
  int id = blockIdx.x * 256 + threadIdx.x;
  int j = id % H;
  int b = (id / H) % B;
  int e = id / (H * B);
  const float* xr = x + (perExpertX ? (size_t)e * B * D : 0) + (size_t)b * D;
  int srow = (perExpertX ? e * B : 0) + b;
  float m = mu[srow], r = rs[srow];
  const float* lg = lng + (size_t)e * D;
  const float* lb = lnb + (size_t)e * D;
  const float* w  = W1 + (size_t)e * D * H + j;
  float acc = 0.f;
  for (int d = 0; d < D; ++d)
    acc += ((xr[d] - m) * r * lg[d] + lb[d]) * w[(size_t)d * H];
  acc += b1[(size_t)e * H + j];
  hbuf[id] = 0.5f * acc * (1.0f + erff(acc * 0.70710678118654752f));  // exact GELU
}

// Expert MLP down: y[e,b,d] = h[e,b,:] @ W2[e] + b2[e]
__global__ __launch_bounds__(256) void k_expert_out(
    const float* __restrict__ hbuf, const float* __restrict__ W2,
    const float* __restrict__ b2, float* __restrict__ y,
    int E, int B, int D, int H)
{
  int id = blockIdx.x * 256 + threadIdx.x;
  int d = id % D;
  int b = (id / D) % B;
  int e = id / (D * B);
  const float* hr = hbuf + ((size_t)e * B + b) * H;
  const float* w  = W2 + (size_t)e * H * D + d;
  float acc = 0.f;
  for (int j = 0; j < H; ++j) acc += hr[j] * w[(size_t)j * D];
  y[id] = acc + b2[(size_t)e * D + d];
}

// Mixture: out[b, ooff+d] = sum_e w[b,e]*y[e,b,d]   (ostride for concat)
__global__ __launch_bounds__(256) void k_mix(
    const float* __restrict__ w, const float* __restrict__ y,
    float* __restrict__ out, int E, int B, int D, int ostride, int ooff)
{
  int id = blockIdx.x * 256 + threadIdx.x;
  int d = id % D, b = id / D;
  float acc = 0.f;
  for (int e = 0; e < E; ++e) acc += w[b * E + e] * y[((size_t)e * B + b) * D + d];
  out[(size_t)b * ostride + ooff + d] = acc;
}

// Column mean over N tokens
__global__ __launch_bounds__(512) void k_colmean(const float* __restrict__ Xp,
                                                 float* __restrict__ Xm) {
  int b = blockIdx.x, d = threadIdx.x;
  const float* p = Xp + (size_t)b * Nn * Dd + d;
  float acc = 0.f;
  for (int n = 0; n < Nn; ++n) acc += p[(size_t)n * Dd];
  Xm[b * Dd + d] = acc * (1.0f / (float)Nn);
}

// Attention logits: one wave32 per (b,n) token row, 4 experts at once
__global__ __launch_bounds__(256) void k_attn_logits(const float* __restrict__ Xp,
                                                     const float* __restrict__ q,
                                                     float* __restrict__ logits) {
  __shared__ float qs[4 * Dd];
  int tid = threadIdx.x;
  for (int i = tid; i < 4 * Dd; i += 256) qs[i] = q[i];
  __syncthreads();
  int wave = tid >> 5, lane = tid & 31;
  int rowi = blockIdx.x * 8 + wave;            // 0 .. 65535
  const float* xr = Xp + (size_t)rowi * Dd;
  float a0 = 0.f, a1 = 0.f, a2 = 0.f, a3 = 0.f;
  for (int i = lane; i < Dd; i += 32) {
    float x = xr[i];
    a0 += x * qs[i];
    a1 += x * qs[Dd + i];
    a2 += x * qs[2 * Dd + i];
    a3 += x * qs[3 * Dd + i];
  }
  for (int off = 16; off > 0; off >>= 1) {
    a0 += __shfl_xor(a0, off, 32);
    a1 += __shfl_xor(a1, off, 32);
    a2 += __shfl_xor(a2, off, 32);
    a3 += __shfl_xor(a3, off, 32);
  }
  if (lane == 0) {
    logits[rowi]              = a0;
    logits[MROWS + rowi]      = a1;
    logits[2 * MROWS + rowi]  = a2;
    logits[3 * MROWS + rowi]  = a3;
  }
}

// In-place row softmax, rows of length L
__global__ __launch_bounds__(256) void k_softmax(float* __restrict__ x, int L) {
  int row = blockIdx.x, tid = threadIdx.x;
  float* xr = x + (size_t)row * L;
  __shared__ float red[256];
  float mx = -1e30f;
  for (int i = tid; i < L; i += 256) mx = fmaxf(mx, xr[i]);
  red[tid] = mx; __syncthreads();
  for (int s = 128; s > 0; s >>= 1) { if (tid < s) red[tid] = fmaxf(red[tid], red[tid + s]); __syncthreads(); }
  mx = red[0]; __syncthreads();
  float sum = 0.f;
  for (int i = tid; i < L; i += 256) { float e = __expf(xr[i] - mx); xr[i] = e; sum += e; }
  red[tid] = sum; __syncthreads();
  for (int s = 128; s > 0; s >>= 1) { if (tid < s) red[tid] += red[tid + s]; __syncthreads(); }
  float inv = 1.0f / red[0];
  for (int i = tid; i < L; i += 256) xr[i] *= inv;
}

// Attention pooling: pooled[e,b,d] += sum_{n in chunk} attn[e,b,n]*Xp[b,n,d]
__global__ __launch_bounds__(512) void k_pool(const float* __restrict__ Xp,
                                              const float* __restrict__ attn,
                                              float* __restrict__ pooled) {
  int b = blockIdx.x >> 3, c = blockIdx.x & 7;
  int d = threadIdx.x;                         // 0..511
  const float* xb = Xp + (size_t)b * Nn * Dd;
  const float* at = attn + (size_t)b * Nn;
  float a0 = 0.f, a1 = 0.f, a2 = 0.f, a3 = 0.f;
  for (int n = c * 512; n < (c + 1) * 512; ++n) {
    float x = xb[(size_t)n * Dd + d];
    a0 += at[n] * x;
    a1 += at[MROWS + n] * x;
    a2 += at[2 * MROWS + n] * x;
    a3 += at[3 * MROWS + n] * x;
  }
  atomicAdd(&pooled[(size_t)0 * Bb * Dd + b * Dd + d], a0);
  atomicAdd(&pooled[(size_t)1 * Bb * Dd + b * Dd + d], a1);
  atomicAdd(&pooled[(size_t)2 * Bb * Dd + b * Dd + d], a2);
  atomicAdd(&pooled[(size_t)3 * Bb * Dd + b * Dd + d], a3);
}

// Head: logits -> sigmoid hazards, survival cumprod
__global__ __launch_bounds__(128) void k_head(const float* __restrict__ FIMMF,
                                              const float* __restrict__ clsW,
                                              const float* __restrict__ clsb,
                                              float* __restrict__ out) {
  int b = blockIdx.x, tid = threadIdx.x;
  __shared__ float red[4 * 128];
  float acc[4] = {0.f, 0.f, 0.f, 0.f};
  for (int d = tid; d < Dd; d += 128) {
    float x = FIMMF[b * Dd + d];
    for (int c = 0; c < 4; ++c) acc[c] += x * clsW[d * 4 + c];
  }
  for (int c = 0; c < 4; ++c) red[c * 128 + tid] = acc[c];
  __syncthreads();
  for (int s = 64; s > 0; s >>= 1) {
    if (tid < s) for (int c = 0; c < 4; ++c) red[c * 128 + tid] += red[c * 128 + tid + s];
    __syncthreads();
  }
  if (tid == 0) {
    float S = 1.0f;
    for (int c = 0; c < 4; ++c) {
      float h = 1.0f / (1.0f + __expf(-(red[c * 128] + clsb[c])));
      out[b * 4 + c] = h;                 // hazards [16,4]
      S *= (1.0f - h);
      out[Bb * 4 + b * 4 + c] = S;        // survival [16,4]
    }
  }
}

// ---------------------------------------------------------------------------
// Host-side orchestration
// ---------------------------------------------------------------------------
extern "C" void kernel_launch(void* const* d_in, const int* in_sizes, int n_in,
                              void* d_out, int out_size, void* d_ws, size_t ws_size,
                              hipStream_t stream) {
  (void)in_sizes; (void)n_in; (void)out_size; (void)ws_size;
  const float* F_gene   = (const float*)d_in[0];
  const float* F_path   = (const float*)d_in[1];
  const float* gene_lng = (const float*)d_in[2];
  const float* gene_lnb = (const float*)d_in[3];
  const float* gene_W   = (const float*)d_in[4];
  const float* gene_b   = (const float*)d_in[5];
  const float* path_lng = (const float*)d_in[6];
  const float* path_lnb = (const float*)d_in[7];
  const float* path_W   = (const float*)d_in[8];
  const float* path_b   = (const float*)d_in[9];
  const float* g_lng = (const float*)d_in[10];
  const float* g_lnb = (const float*)d_in[11];
  const float* g_W1  = (const float*)d_in[12];
  const float* g_b1  = (const float*)d_in[13];
  const float* g_W2  = (const float*)d_in[14];
  const float* g_b2  = (const float*)d_in[15];
  const float* p_q   = (const float*)d_in[16];
  const float* p_lng = (const float*)d_in[17];
  const float* p_lnb = (const float*)d_in[18];
  const float* p_W1  = (const float*)d_in[19];
  const float* p_b1  = (const float*)d_in[20];
  const float* p_W2  = (const float*)d_in[21];
  const float* p_b2  = (const float*)d_in[22];
  const float* f_lng = (const float*)d_in[23];
  const float* f_lnb = (const float*)d_in[24];
  const float* f_W1  = (const float*)d_in[25];
  const float* f_b1  = (const float*)d_in[26];
  const float* f_W2  = (const float*)d_in[27];
  const float* f_b2  = (const float*)d_in[28];
  const float* gg_lng = (const float*)d_in[29];
  const float* gg_lnb = (const float*)d_in[30];
  const float* gg_W   = (const float*)d_in[31];
  const float* gg_b   = (const float*)d_in[32];
  const float* pg_lng = (const float*)d_in[33];
  const float* pg_lnb = (const float*)d_in[34];
  const float* pg_W   = (const float*)d_in[35];
  const float* pg_b   = (const float*)d_in[36];
  const float* fg_lng = (const float*)d_in[37];
  const float* fg_lnb = (const float*)d_in[38];
  const float* fg_W   = (const float*)d_in[39];
  const float* fg_b   = (const float*)d_in[40];
  const float* fin_lng = (const float*)d_in[41];
  const float* fin_lnb = (const float*)d_in[42];
  const float* fin_W   = (const float*)d_in[43];
  const float* fin_b   = (const float*)d_in[44];
  const float* cls_W   = (const float*)d_in[45];
  const float* cls_b   = (const float*)d_in[46];
  float* out = (float*)d_out;

  // Workspace carve-out (~140 MB)
  char* wsb = (char*)d_ws;
  size_t off = 0;
  auto alloc = [&](size_t bytes) -> char* {
    size_t a = (off + 255) & ~(size_t)255;
    off = a + bytes;
    return wsb + a;
  };
  _Float16* WhT = (_Float16*)alloc((size_t)Dd * PIN_ * 2);
  float* muP = (float*)alloc((size_t)MROWS * 4);
  float* rsP = (float*)alloc((size_t)MROWS * 4);
  float* Xp  = (float*)alloc((size_t)MROWS * Dd * 4);       // 128 MB
  float* muG = (float*)alloc(Bb * 4);
  float* rsG = (float*)alloc(Bb * 4);
  float* Xg  = (float*)alloc((size_t)Bb * Dd * 4);
  float* muXg = (float*)alloc(Bb * 4);
  float* rsXg = (float*)alloc(Bb * 4);
  float* ghbuf = (float*)alloc((size_t)4 * Bb * Hh * 4);
  float* gout  = (float*)alloc((size_t)4 * Bb * Dd * 4);
  float* wg    = (float*)alloc(Bb * 4 * 4);
  float* attn  = (float*)alloc((size_t)4 * MROWS * 4);      // logits -> attn
  float* Xm    = (float*)alloc((size_t)Bb * Dd * 4);
  float* muXm  = (float*)alloc(Bb * 4);
  float* rsXm  = (float*)alloc(Bb * 4);
  float* wp    = (float*)alloc(Bb * 4 * 4);
  float* pooled = (float*)alloc((size_t)4 * Bb * Dd * 4);
  float* muPl  = (float*)alloc(4 * Bb * 4);
  float* rsPl  = (float*)alloc(4 * Bb * 4);
  float* phbuf = (float*)alloc((size_t)4 * Bb * Hh * 4);
  float* pout  = (float*)alloc((size_t)4 * Bb * Dd * 4);
  float* fused = (float*)alloc((size_t)Bb * 2 * Dd * 4);
  float* muF   = (float*)alloc(Bb * 4);
  float* rsF   = (float*)alloc(Bb * 4);
  float* wf    = (float*)alloc(Bb * 2 * 4);
  float* fin   = (float*)alloc((size_t)Bb * Dd * 4);
  float* muFi  = (float*)alloc(Bb * 4);
  float* rsFi  = (float*)alloc(Bb * 4);
  float* fhbuf = (float*)alloc((size_t)2 * Bb * Hh * 4);
  float* fout  = (float*)alloc((size_t)2 * Bb * Dd * 4);
  float* FIMMF = (float*)alloc((size_t)Bb * Dd * 4);

  // ---- path branch: LN stats -> f16 weight transpose -> WMMA GEMM ----
  k_cvt_w<<<(Dd * PIN_) / 256, 256, 0, stream>>>(path_W, WhT);
  k_rowstats<<<MROWS, 256, 0, stream>>>(F_path, PIN_, muP, rsP);
  k_path_gemm<<<dim3(Dd / BN, MROWS / BM), 256, 0, stream>>>(
      F_path, muP, rsP, path_lng, path_lnb, WhT, path_b, Xp);

  // ---- gene branch ----
  k_rowstats<<<Bb, 256, 0, stream>>>(F_gene, GIN_, muG, rsG);
  k_ln_proj<<<(Bb * Dd) / 256, 256, 0, stream>>>(F_gene, muG, rsG, gene_lng,
                                                 gene_lnb, gene_W, gene_b, Xg,
                                                 GIN_, Dd);
  k_rowstats<<<Bb, 256, 0, stream>>>(Xg, Dd, muXg, rsXg);
  k_gate<<<Bb, 256, 0, stream>>>(Xg, muXg, rsXg, gg_lng, gg_lnb, gg_W, gg_b, wg, Dd, 4);
  k_expert_h<<<(4 * Bb * Hh) / 256, 256, 0, stream>>>(Xg, muXg, rsXg, g_lng, g_lnb,
                                                      g_W1, g_b1, ghbuf, 4, Bb, Dd, Hh, 0);
  k_expert_out<<<(4 * Bb * Dd) / 256, 256, 0, stream>>>(ghbuf, g_W2, g_b2, gout, 4, Bb, Dd, Hh);
  k_mix<<<(Bb * Dd) / 256, 256, 0, stream>>>(wg, gout, fused, 4, Bb, Dd, 2 * Dd, 0);

  // ---- path experts: mean, gate, attention pooling ----
  k_colmean<<<Bb, Dd, 0, stream>>>(Xp, Xm);
  k_rowstats<<<Bb, 256, 0, stream>>>(Xm, Dd, muXm, rsXm);
  k_gate<<<Bb, 256, 0, stream>>>(Xm, muXm, rsXm, pg_lng, pg_lnb, pg_W, pg_b, wp, Dd, 4);
  k_attn_logits<<<MROWS / 8, 256, 0, stream>>>(Xp, p_q, attn);
  k_softmax<<<4 * Bb, 256, 0, stream>>>(attn, Nn);
  hipMemsetAsync(pooled, 0, (size_t)4 * Bb * Dd * 4, stream);
  k_pool<<<Bb * 8, 512, 0, stream>>>(Xp, attn, pooled);
  k_rowstats<<<4 * Bb, 256, 0, stream>>>(pooled, Dd, muPl, rsPl);
  k_expert_h<<<(4 * Bb * Hh) / 256, 256, 0, stream>>>(pooled, muPl, rsPl, p_lng, p_lnb,
                                                      p_W1, p_b1, phbuf, 4, Bb, Dd, Hh, 1);
  k_expert_out<<<(4 * Bb * Dd) / 256, 256, 0, stream>>>(phbuf, p_W2, p_b2, pout, 4, Bb, Dd, Hh);
  k_mix<<<(Bb * Dd) / 256, 256, 0, stream>>>(wp, pout, fused, 4, Bb, Dd, 2 * Dd, Dd);

  // ---- fusion experts ----
  k_rowstats<<<Bb, 256, 0, stream>>>(fused, 2 * Dd, muF, rsF);
  k_gate<<<Bb, 256, 0, stream>>>(fused, muF, rsF, fg_lng, fg_lnb, fg_W, fg_b, wf, 2 * Dd, 2);
  k_ln_proj<<<(Bb * Dd) / 256, 256, 0, stream>>>(fused, muF, rsF, fin_lng, fin_lnb,
                                                 fin_W, fin_b, fin, 2 * Dd, Dd);
  k_rowstats<<<Bb, 256, 0, stream>>>(fin, Dd, muFi, rsFi);
  k_expert_h<<<(2 * Bb * Hh) / 256, 256, 0, stream>>>(fin, muFi, rsFi, f_lng, f_lnb,
                                                      f_W1, f_b1, fhbuf, 2, Bb, Dd, Hh, 0);
  k_expert_out<<<(2 * Bb * Dd) / 256, 256, 0, stream>>>(fhbuf, f_W2, f_b2, fout, 2, Bb, Dd, Hh);
  k_mix<<<(Bb * Dd) / 256, 256, 0, stream>>>(wf, fout, FIMMF, 2, Bb, Dd, Dd, 0);

  // ---- head: hazards + survival ----
  k_head<<<Bb, 128, 0, stream>>>(FIMMF, cls_W, cls_b, out);
}
